// HINGCN_GS_77103252897854
// MI455X (gfx1250) — compile-verified, matrix operand order checked by reading, not run on hardware
//
#include <hip/hip_runtime.h>
#include <hip/hip_bf16.h>
#include <math.h>

typedef _Float16 v16h __attribute__((ext_vector_type(16)));
typedef _Float16 v8h  __attribute__((ext_vector_type(8)));
typedef float    v8f  __attribute__((ext_vector_type(8)));

// Problem constants (from reference setup_inputs)
#define B    512
#define S    10
#define D    256
#define E    64
#define KCAT 576      // 2*D + E
#define BS   (B * S)  // 5120
#define C_OUT 8

// ---------------- workspace layout (bytes) ----------------
static const size_t OFF_WCAT  = 0;                               // 4 * 256*576 f16 (transposed: [n][k])
static const size_t OFF_WE    = OFF_WCAT  + (size_t)4*D*KCAT*2;  // 4 * 64*576 f16 (transposed)
static const size_t OFF_ABIG  = OFF_WE    + (size_t)4*E*KCAT*2;  // 5120*576 f16 (A1 / EA reuse)
static const size_t OFF_A0    = OFF_ABIG  + (size_t)BS*KCAT*2;   // 512*576 f16 (A0 / A0p reuse)
static const size_t OFF_G1    = OFF_A0    + (size_t)B*KCAT*2;    // 5120*256 f16
static const size_t OFF_G0    = OFF_G1    + (size_t)BS*D*2;      // 512*256 f16
static const size_t OFF_E0P   = OFF_G0    + (size_t)B*D*2;       // 5120*64 f16
static const size_t OFF_OUTMP = OFF_E0P   + (size_t)BS*E*2;      // 2*512*256 f32

// ---------------- weight packing (transposed to [n][k] for fragment-friendly LDS tiles) ----
__global__ __launch_bounds__(256) void build_wcat_t(
    const float* __restrict__ Ws, const float* __restrict__ Wn,
    const float* __restrict__ Wedge, _Float16* __restrict__ out)
{
    int i = blockIdx.x * 256 + threadIdx.x;
    const int total = 4 * D * KCAT;
    if (i >= total) return;
    int k  = i % KCAT;
    int n  = (i / KCAT) & (D - 1);
    int ml = i / (KCAT * D);
    float v;
    if (k < D)            v = Ws[((size_t)ml * D + k) * D + n];
    else if (k < 2 * D)   v = Wn[((size_t)ml * D + (k - D)) * D + n];
    else                  v = Wedge[((size_t)ml * E + (k - 2 * D)) * D + n];
    out[i] = (_Float16)v;
}

// WeT[ml][n][k] = We[ml][k][n]  (We: [2,2,576,64])
__global__ __launch_bounds__(256) void build_wet_t(
    const float* __restrict__ We, _Float16* __restrict__ out)
{
    int i = blockIdx.x * 256 + threadIdx.x;
    const int total = 4 * E * KCAT;
    if (i >= total) return;
    int k  = i % KCAT;
    int n  = (i / KCAT) & (E - 1);
    int ml = i / (KCAT * E);
    out[i] = (_Float16)We[((size_t)ml * KCAT + k) * E + n];
}

// ---------------- gather + mean builders ----------------
__global__ __launch_bounds__(256) void build_A1(
    const float* __restrict__ feats, const float* __restrict__ emb,
    const int* __restrict__ n0, const int* __restrict__ n1,
    const int* __restrict__ e1, _Float16* __restrict__ A)
{
    const int r = blockIdx.x;          // 0..BS-1
    const int t = threadIdx.x;         // 0..255
    _Float16* row = A + (size_t)r * KCAT;
    row[t] = (_Float16)feats[(size_t)n0[r] * D + t];
    float acc = 0.f;
#pragma unroll
    for (int s = 0; s < S; ++s) acc += feats[(size_t)n1[r * S + s] * D + t];
    row[D + t] = (_Float16)(acc * 0.1f);
    if (t < E) {
        float ea = 0.f;
#pragma unroll
        for (int s = 0; s < S; ++s) ea += emb[(size_t)e1[r * S + s] * E + t];
        row[2 * D + t] = (_Float16)(ea * 0.1f);
    }
}

__global__ __launch_bounds__(256) void build_A0(
    const float* __restrict__ feats, const float* __restrict__ emb,
    const int* __restrict__ ids, const int* __restrict__ n0,
    const int* __restrict__ e0, _Float16* __restrict__ A)
{
    const int r = blockIdx.x;          // 0..B-1
    const int t = threadIdx.x;
    _Float16* row = A + (size_t)r * KCAT;
    row[t] = (_Float16)feats[(size_t)ids[r] * D + t];
    float acc = 0.f;
#pragma unroll
    for (int s = 0; s < S; ++s) acc += feats[(size_t)n0[r * S + s] * D + t];
    row[D + t] = (_Float16)(acc * 0.1f);
    if (t < E) {
        float ea = 0.f;
#pragma unroll
        for (int s = 0; s < S; ++s) ea += emb[(size_t)e0[r * S + s] * E + t];
        row[2 * D + t] = (_Float16)(ea * 0.1f);
    }
}

// EA[r] = [ g0[r/S] | g1[r] | f16(emb[e0[r]]) ]
__global__ __launch_bounds__(256) void build_EA(
    const _Float16* __restrict__ g0h, const _Float16* __restrict__ g1h,
    const float* __restrict__ emb, const int* __restrict__ e0,
    _Float16* __restrict__ A)
{
    const int r = blockIdx.x;          // 0..BS-1
    const int t = threadIdx.x;
    _Float16* row = A + (size_t)r * KCAT;
    row[t]     = g0h[(size_t)(r / S) * D + t];
    row[D + t] = g1h[(size_t)r * D + t];
    if (t < E) row[2 * D + t] = (_Float16)emb[(size_t)e0[r] * E + t];
}

// A0p[r] = [ g0[r] | mean_s g1[r*S+s] | mean_s e0'[r*S+s] ]
__global__ __launch_bounds__(256) void build_A0p(
    const _Float16* __restrict__ g0h, const _Float16* __restrict__ g1h,
    const _Float16* __restrict__ e0ph, _Float16* __restrict__ A)
{
    const int r = blockIdx.x;          // 0..B-1
    const int t = threadIdx.x;
    _Float16* row = A + (size_t)r * KCAT;
    row[t] = g0h[(size_t)r * D + t];
    float acc = 0.f;
#pragma unroll
    for (int s = 0; s < S; ++s) acc += (float)g1h[(size_t)(r * S + s) * D + t];
    row[D + t] = (_Float16)(acc * 0.1f);
    if (t < E) {
        float ea = 0.f;
#pragma unroll
        for (int s = 0; s < S; ++s) ea += (float)e0ph[(size_t)(r * S + s) * E + t];
        row[2 * D + t] = (_Float16)(ea * 0.1f);
    }
}

// ---------------- WMMA GEMM (double-buffered LDS, register prefetch, peeled epilogue) ----
// C[M,N] = act(A[M,K] @ W^T)  A row-major f16 (lda=K), Wt transposed f16 [N][K],
// K multiple of 32 (>= 64). Block tile 32(M) x 64(N), 8 waves, 1x 16x16 wmma tile each.
// mode 0: relu -> f16; mode 1: +bias, tanh -> f16; mode 2: raw f32
__global__ __launch_bounds__(256) void gemm_wmma(
    const _Float16* __restrict__ A, const _Float16* __restrict__ Wt,
    void* __restrict__ Cout, int K, int N, int mode,
    const float* __restrict__ bias)
{
    // padded row stride 40 halves = 80 B: every 8-half chunk stays 16B-aligned,
    // 20-dword stride spreads LDS banks. Two buffers for global/compute overlap.
    __shared__ __align__(16) _Float16 As[2][32][40];   // [buf][m][k]
    __shared__ __align__(16) _Float16 Bs[2][64][40];   // [buf][n][k]

    const int tid  = threadIdx.x;
    const int lane = tid & 31;
    const int wave = tid >> 5;
    const int wm   = wave >> 2;                 // 0..1  (M sub-tile)
    const int wn   = wave & 3;                  // 0..3  (N sub-tile)
    const int m0   = blockIdx.x * 32;
    const int n0   = blockIdx.y * 64;

    const int  laneM = lane & 15;
    const bool hi    = lane >= 16;
    const int  kOff  = hi ? 8 : 0;              // ISA 16-bit A/B fragment K sub-block

    // per-thread staging source/dest coordinates
    const int ar = tid >> 3, ac = (tid & 7) * 4;   // A: 4 halves (8B) each
    const int br = tid >> 2, bc = (tid & 3) * 8;   // B: 8 halves (16B) each
    const _Float16* aSrc = A  + (size_t)(m0 + ar) * K + ac;
    const _Float16* bSrc = Wt + (size_t)(n0 + br) * K + bc;

    v8f acc = {};

    // prefetch tile 0 into registers
    uint2 ra = *(const uint2*)(aSrc);
    uint4 rb = *(const uint4*)(bSrc);

    int buf = 0;
    // main loop: unconditional next-tile load (last tile handled in epilogue)
    for (int k0 = 0; k0 < K - 32; k0 += 32) {
        *(uint2*)&As[buf][ar][ac] = ra;
        *(uint4*)&Bs[buf][br][bc] = rb;
        __syncthreads();

        ra = *(const uint2*)(aSrc + k0 + 32);
        rb = *(const uint4*)(bSrc + k0 + 32);
        // WGP-scope prefetch (locality 3 -> pull into all cache levels);
        // speculative, safe if it runs past the tile.
        __builtin_prefetch(aSrc + k0 + 64, 0, 3);
        __builtin_prefetch(bSrc + k0 + 64, 0, 3);

        const v8h a_lo = *(const v8h*)&As[buf][wm * 16 + laneM][kOff];
        const v8h a_hi = *(const v8h*)&As[buf][wm * 16 + laneM][16 + kOff];
        const v8h b_lo = *(const v8h*)&Bs[buf][wn * 16 + laneM][kOff];
        const v8h b_hi = *(const v8h*)&Bs[buf][wn * 16 + laneM][16 + kOff];
        const v16h af = __builtin_shufflevector(a_lo, a_hi,
            0, 1, 2, 3, 4, 5, 6, 7, 8, 9, 10, 11, 12, 13, 14, 15);
        const v16h bf = __builtin_shufflevector(b_lo, b_hi,
            0, 1, 2, 3, 4, 5, 6, 7, 8, 9, 10, 11, 12, 13, 14, 15);

        acc = __builtin_amdgcn_wmma_f32_16x16x32_f16(
                  false, af, false, bf, (short)0, acc, false, false);
        buf ^= 1;
    }

    // epilogue: commit and consume the final tile
    {
        *(uint2*)&As[buf][ar][ac] = ra;
        *(uint4*)&Bs[buf][br][bc] = rb;
        __syncthreads();

        const v8h a_lo = *(const v8h*)&As[buf][wm * 16 + laneM][kOff];
        const v8h a_hi = *(const v8h*)&As[buf][wm * 16 + laneM][16 + kOff];
        const v8h b_lo = *(const v8h*)&Bs[buf][wn * 16 + laneM][kOff];
        const v8h b_hi = *(const v8h*)&Bs[buf][wn * 16 + laneM][16 + kOff];
        const v16h af = __builtin_shufflevector(a_lo, a_hi,
            0, 1, 2, 3, 4, 5, 6, 7, 8, 9, 10, 11, 12, 13, 14, 15);
        const v16h bf = __builtin_shufflevector(b_lo, b_hi,
            0, 1, 2, 3, 4, 5, 6, 7, 8, 9, 10, 11, 12, 13, 14, 15);

        acc = __builtin_amdgcn_wmma_f32_16x16x32_f16(
                  false, af, false, bf, (short)0, acc, false, false);
    }

    // C/D layout: VGPR v -> M = base + v, lanes>=16 -> M += 8, N = lane&15
    const int cm = m0 + wm * 16 + (hi ? 8 : 0);
    const int cn = n0 + wn * 16 + laneM;
    if (mode == 2) {
        float* C = (float*)Cout;
#pragma unroll
        for (int v = 0; v < 8; ++v) C[(size_t)(cm + v) * N + cn] = acc[v];
    } else if (mode == 0) {
        _Float16* C = (_Float16*)Cout;
#pragma unroll
        for (int v = 0; v < 8; ++v) {
            float x = acc[v];
            C[(size_t)(cm + v) * N + cn] = (_Float16)(x > 0.f ? x : 0.f);
        }
    } else {
        _Float16* C = (_Float16*)Cout;
        const float bb = bias[cn];
#pragma unroll
        for (int v = 0; v < 8; ++v)
            C[(size_t)(cm + v) * N + cn] = (_Float16)tanhf(acc[v] + bb);
    }
}

// ---------------- final: metapath mean, row-normalize, FC ----------------
__global__ __launch_bounds__(256) void final_fc(
    const float* __restrict__ outmp, const float* __restrict__ fc_w,
    const float* __restrict__ fc_b, float* __restrict__ out)
{
    __shared__ float sh[256];
    __shared__ float ph[256 * C_OUT];
    const int r = blockIdx.x;          // 0..B-1
    const int t = threadIdx.x;         // 0..255 (one per D channel)
    const float v = 0.5f * (outmp[(size_t)r * D + t] +
                            outmp[(size_t)(B + r) * D + t]);
    sh[t] = v * v;
    __syncthreads();
    for (int off = 128; off > 0; off >>= 1) {
        if (t < off) sh[t] += sh[t + off];
        __syncthreads();
    }
    const float nv = v / fmaxf(sqrtf(sh[0]), 1e-12f);
#pragma unroll
    for (int c = 0; c < C_OUT; ++c) ph[t * C_OUT + c] = nv * fc_w[t * C_OUT + c];
    __syncthreads();
    if (t < C_OUT) {
        float s = fc_b[t];
        for (int i = 0; i < 256; ++i) s += ph[i * C_OUT + t];
        out[r * C_OUT + t] = s;
    }
}

// ---------------- driver ----------------
extern "C" void kernel_launch(void* const* d_in, const int* in_sizes, int n_in,
                              void* d_out, int out_size, void* d_ws, size_t ws_size,
                              hipStream_t stream)
{
    const int*   ids    = (const int*)d_in[0];
    const float* feats  = (const float*)d_in[1];
    const int*   nA[2]  = { (const int*)d_in[2], (const int*)d_in[4] };   // n00, n10
    const int*   nB[2]  = { (const int*)d_in[3], (const int*)d_in[5] };   // n01, n11
    const int*   eA_[2] = { (const int*)d_in[6], (const int*)d_in[8] };   // e00, e10
    const int*   eB_[2] = { (const int*)d_in[7], (const int*)d_in[9] };   // e01, e11
    const float* emb[2] = { (const float*)d_in[10], (const float*)d_in[11] };
    const float* W_self  = (const float*)d_in[12];
    const float* W_neigh = (const float*)d_in[13];
    const float* W_edge  = (const float*)d_in[14];
    const float* We      = (const float*)d_in[15];   // [2,2,576,64]
    const float* be      = (const float*)d_in[16];   // [2,2,64]
    const float* fc_w    = (const float*)d_in[17];
    const float* fc_b    = (const float*)d_in[18];
    float* out = (float*)d_out;

    char* ws = (char*)d_ws;
    _Float16* WcatT = (_Float16*)(ws + OFF_WCAT);
    _Float16* WeT   = (_Float16*)(ws + OFF_WE);
    _Float16* Abig  = (_Float16*)(ws + OFF_ABIG);
    _Float16* A0buf = (_Float16*)(ws + OFF_A0);
    _Float16* g1h   = (_Float16*)(ws + OFF_G1);
    _Float16* g0h   = (_Float16*)(ws + OFF_G0);
    _Float16* e0ph  = (_Float16*)(ws + OFF_E0P);
    float*    outmp = (float*)(ws + OFF_OUTMP);

    // pack weights to f16, transposed [n][k], concatenated K = [self | neigh | edge]
    build_wcat_t<<<(4 * D * KCAT + 255) / 256, 256, 0, stream>>>(W_self, W_neigh, W_edge, WcatT);
    build_wet_t<<<(4 * E * KCAT + 255) / 256, 256, 0, stream>>>(We, WeT);

    for (int mp = 0; mp < 2; ++mp) {
        const _Float16* Wc0 = WcatT + (size_t)(mp * 2 + 0) * D * KCAT;
        const _Float16* Wc1 = WcatT + (size_t)(mp * 2 + 1) * D * KCAT;
        const _Float16* Wee = WeT   + (size_t)(mp * 2 + 0) * E * KCAT;
        const float*    be0 = be    + (size_t)(mp * 2 + 0) * E;

        // layer 0, k=1: g1 = relu(A1 @ Wcat0)   [5120,256]
        build_A1<<<BS, 256, 0, stream>>>(feats, emb[mp], nA[mp], nB[mp], eB_[mp], Abig);
        gemm_wmma<<<dim3(BS / 32, D / 64), 256, 0, stream>>>(Abig, Wc0, g1h, KCAT, D, 0, nullptr);

        // layer 0, k=0: g0 = relu(A0 @ Wcat0)   [512,256]
        build_A0<<<B, 256, 0, stream>>>(feats, emb[mp], ids, nA[mp], eA_[mp], A0buf);
        gemm_wmma<<<dim3(B / 32, D / 64), 256, 0, stream>>>(A0buf, Wc0, g0h, KCAT, D, 0, nullptr);

        // edge update: e0' = tanh([rep(g0)|g1|e0g] @ We0 + be0)   [5120,64]
        build_EA<<<BS, 256, 0, stream>>>(g0h, g1h, emb[mp], eA_[mp], Abig);
        gemm_wmma<<<dim3(BS / 32, E / 64), 256, 0, stream>>>(Abig, Wee, e0ph, KCAT, E, 1, be0);

        // layer 1: out_mp = A0p @ Wcat1   [512,256] f32
        build_A0p<<<B, 256, 0, stream>>>(g0h, g1h, e0ph, A0buf);
        gemm_wmma<<<dim3(B / 32, D / 64), 256, 0, stream>>>(
            A0buf, Wc1, outmp + (size_t)mp * B * D, KCAT, D, 2, nullptr);
    }

    final_fc<<<B, 256, 0, stream>>>(outmp, fc_w, fc_b, out);
}
